// LocalDenseConv2D_73658689126882
// MI455X (gfx1250) — matrix operand order repaired and checked by reading, not compiled, not run on hardware
//
#include <hip/hip_runtime.h>

typedef __attribute__((ext_vector_type(16))) _Float16 v16h;
typedef __attribute__((ext_vector_type(8)))  _Float16 v8h;
typedef __attribute__((ext_vector_type(8)))  float    v8f;
typedef __attribute__((ext_vector_type(4)))  float    f32x4;

// Problem dims
#define BB    8
#define INC   64
#define INL   64
#define TT    256
#define OUTC  64
#define OUTL  64
#define LP    66           // INL + 2 (padded rows)
#define TP2   264          // TT + 2 halo, padded to 16B-aligned rows (264*2B = 33*16B)
#define TLS   132          // staged t extent in LDS (130 used, 132 alloc)
#define CSTR  72           // cin stride in LDS (72*2B = 9*16B: aligned + bank spread)

// ---------------------------------------------------------------------------
// Kernel 1: pad + convert x (f32) -> xh (f16), layout [B][INC][LP][TP2],
// zero halo at tp==0 and tp>=257, lp==0 and lp==65.
// ---------------------------------------------------------------------------
__global__ __launch_bounds__(256) void k_pad_x(const float* __restrict__ x,
                                               _Float16* __restrict__ xh) {
    int idx = blockIdx.x * 256 + threadIdx.x;           // total = B*INC*LP*TP2
    if (idx >= BB * INC * LP * TP2) return;
    int tp = idx % TP2;
    int r  = idx / TP2;
    int lp = r % LP;  r /= LP;
    int c  = r % INC;
    int b  = r / INC;
    float v = 0.0f;
    if (tp >= 1 && tp <= TT && lp >= 1 && lp <= INL)
        v = x[((size_t)(b * INC + c) * INL + (lp - 1)) * TT + (tp - 1)];
    xh[idx] = (_Float16)v;
}

// ---------------------------------------------------------------------------
// Kernel 2: weight (K=cin*9+r, OUTC, OUTL) f32 -> wh[l][r][c][cin] f16
// (cin contiguous) so a B-fragment lane (N = c) reads 16 consecutive f16.
// ---------------------------------------------------------------------------
__global__ __launch_bounds__(256) void k_conv_w(const float* __restrict__ w,
                                                _Float16* __restrict__ wh) {
    int idx = blockIdx.x * 256 + threadIdx.x;           // total = 64*9*64*64
    if (idx >= OUTL * 9 * OUTC * INC) return;
    int cin = idx & 63;
    int r2  = idx >> 6;
    int c   = r2 & 63;  r2 >>= 6;
    int r   = r2 % 9;
    int l   = r2 / 9;
    wh[idx] = (_Float16)w[((size_t)(cin * 9 + r) * OUTC + c) * OUTL + l];
}

// ---------------------------------------------------------------------------
// Kernel 3: locally-connected conv as 9 shifted GEMMs (K = cin) via
// v_wmma_f32_16x16x32_f16.  One workgroup (8 waves) per (l, b, 128-t block):
// stage x halo tile into LDS transposed to cin-contiguous, then each wave
// computes a 16(t) x 64(c) strip.  A-fragments = 2 aligned ds_load_b128.
// ---------------------------------------------------------------------------
__global__ __launch_bounds__(256) void k_gemm(const _Float16* __restrict__ xh,
                                              const _Float16* __restrict__ wh,
                                              const float* __restrict__ bias,
                                              float* __restrict__ conv) {
    __shared__ _Float16 xs[3 * TLS * CSTR];             // 57,024 B

    const int l   = blockIdx.x;                         // 0..63
    const int bt  = blockIdx.y;                         // 0..15
    const int b   = bt >> 1;
    const int t0b = (bt & 1) << 7;                      // 0 or 128
    const int tid = threadIdx.x;

    // ---- stage: global [cin][row][t-contig] -> LDS [row][t][cin-contig] ----
    const _Float16* xg = xh + ((size_t)b * INC * LP + (size_t)l) * TP2 + t0b;
    for (int s = tid; s < INC * 3 * 17; s += 256) {     // 17 chunks of 8 t
        int tc  = s % 17;
        int r2  = s / 17;
        int row = r2 % 3;
        int cin = r2 / 3;
        v8h vv = *(const v8h*)(xg + ((size_t)cin * LP + row) * TP2 + tc * 8);
#pragma unroll
        for (int e = 0; e < 8; ++e) {
            int tl = tc * 8 + e;
            if (tl < TLS) xs[(row * TLS + tl) * CSTR + cin] = vv[e];
        }
    }
    __syncthreads();

    const int w8    = tid >> 5;                         // wave 0..7 -> t-tile
    const int lane  = tid & 31;
    const int laneM = lane & 15;                        // A: M(t); B/D: N(c)
    const int hi    = lane >> 4;

    v8f acc[4] = {};
    const _Float16* wl = wh + (size_t)l * 9 * OUTC * INC;

#pragma unroll
    for (int r = 0; r < 9; ++r) {                       // (i,j) compile-time
        const int i = r / 3, j = r % 3;
#pragma unroll
        for (int cc = 0; cc < 2; ++cc) {                // cin 0..31 / 32..63
            // A fragment: lanes 0-15 K=0..15; lanes 16-31 K=16..31.
            const _Float16* ap =
                &xs[(i * TLS + (w8 * 16 + laneM + j)) * CSTR + cc * 32 + hi * 16];
            union { v16h v; v8h h[2]; } au;
            au.h[0] = *(const v8h*)ap;                  // 16B aligned
            au.h[1] = *(const v8h*)(ap + 8);            // +16B
#pragma unroll
            for (int nt = 0; nt < 4; ++nt) {
                // B fragment: lane N=laneM, 16 consecutive cin (32B aligned)
                const _Float16* wp = wl
                    + ((size_t)(r * OUTC + nt * 16 + laneM)) * INC
                    + cc * 32 + hi * 16;
                v16h bfrag = *(const v16h*)wp;
                acc[nt] = __builtin_amdgcn_wmma_f32_16x16x32_f16(
                    false, au.v, false, bfrag, (short)0, acc[nt], false, false);
            }
        }
    }

    // D layout: VGPR v holds M=v (+8 for hi half), N=laneM -> per lane 8
    // consecutive t values: one aligned 32B store, bias fused.
#pragma unroll
    for (int nt = 0; nt < 4; ++nt) {
        int c = nt * 16 + laneM;
        float bv = bias[c * OUTL + l];
        v8f o = acc[nt];
#pragma unroll
        for (int v = 0; v < 8; ++v) o[v] += bv;
        float* outp = conv + (((size_t)(b * OUTC + c) * OUTL + l) * TT
                              + t0b + w8 * 16 + hi * 8);
        *(v8f*)outp = o;
    }
}

// ---------------------------------------------------------------------------
// Kernel 4: per-channel batch statistics -> fused scale/shift.
// ---------------------------------------------------------------------------
__global__ __launch_bounds__(256) void k_stats(const float* __restrict__ conv,
                                               const float* __restrict__ gamma,
                                               const float* __restrict__ beta,
                                               float* __restrict__ stats) {
    __shared__ float ssum[256];
    __shared__ float ssq[256];
    const int ch  = blockIdx.x;
    const int tid = threadIdx.x;
    float s = 0.f, q = 0.f;
    for (int n = tid; n < BB * OUTL * TT; n += 256) {
        int b   = n >> 14;                              // OUTL*TT = 16384
        int off = n & 16383;
        float v = conv[(((size_t)b * OUTC + ch) << 14) + off];
        s += v;
        q += v * v;
    }
    ssum[tid] = s; ssq[tid] = q;
    __syncthreads();
    for (int st = 128; st > 0; st >>= 1) {
        if (tid < st) { ssum[tid] += ssum[tid + st]; ssq[tid] += ssq[tid + st]; }
        __syncthreads();
    }
    if (tid == 0) {
        const float invN = 1.0f / (float)(BB * OUTL * TT);
        float mean = ssum[0] * invN;
        float var  = ssq[0] * invN - mean * mean;
        float sc   = gamma[ch] * rsqrtf(var + 1e-5f);
        stats[ch * 2 + 0] = sc;
        stats[ch * 2 + 1] = beta[ch] - mean * sc;
    }
}

// ---------------------------------------------------------------------------
// Kernel 5: BN (scale/shift) + PReLU, float4 vectorized.
// ---------------------------------------------------------------------------
__global__ __launch_bounds__(256) void k_bn_prelu(const float* __restrict__ conv,
                                                  const float* __restrict__ stats,
                                                  const float* __restrict__ alpha,
                                                  float* __restrict__ out) {
    int i4 = blockIdx.x * 256 + threadIdx.x;
    size_t base = (size_t)i4 * 4;
    int ch = (int)((base >> 14) & (OUTC - 1));
    float sc = stats[ch * 2 + 0];
    float sh = stats[ch * 2 + 1];
    float al = alpha[0];
    f32x4 v = ((const f32x4*)conv)[i4];
#pragma unroll
    for (int e = 0; e < 4; ++e) {
        float y = v[e] * sc + sh;
        v[e] = (y >= 0.f) ? y : al * y;
    }
    ((f32x4*)out)[i4] = v;
}

// ---------------------------------------------------------------------------
// Launcher
// ---------------------------------------------------------------------------
extern "C" void kernel_launch(void* const* d_in, const int* in_sizes, int n_in,
                              void* d_out, int out_size, void* d_ws, size_t ws_size,
                              hipStream_t stream) {
    (void)in_sizes; (void)n_in; (void)out_size; (void)ws_size;
    const float* x     = (const float*)d_in[0];
    const float* w     = (const float*)d_in[1];
    const float* bias  = (const float*)d_in[2];
    const float* gamma = (const float*)d_in[3];
    const float* beta  = (const float*)d_in[4];
    const float* alpha = (const float*)d_in[5];
    float* out = (float*)d_out;

    // Workspace layout (bytes):
    //   xh   : B*INC*LP*TP2 * 2  = 17,842,176  (+256 slack)
    //   wh   : 64*9*64*64 * 2    =  4,718,592
    //   conv : B*OUTC*OUTL*TT*4  = 33,554,432
    //   stats: OUTC * 2 * 4      =        512     (total ~56.1 MB)
    char* ws = (char*)d_ws;
    _Float16* xh    = (_Float16*)(ws);
    _Float16* wh    = (_Float16*)(ws + 17842432);
    float*    conv  = (float*)   (ws + 22561024);
    float*    stats = (float*)   (ws + 56115456);

    const int nPad = BB * INC * LP * TP2;               // 8,921,088
    k_pad_x<<<(nPad + 255) / 256, 256, 0, stream>>>(x, xh);

    const int nW = OUTL * 9 * OUTC * INC;               // 2,359,296
    k_conv_w<<<(nW + 255) / 256, 256, 0, stream>>>(w, wh);

    dim3 gg(OUTL, BB * 2);                              // (64, 16) blocks x 256
    k_gemm<<<gg, 256, 0, stream>>>(xh, wh, bias, conv);

    k_stats<<<OUTC, 256, 0, stream>>>(conv, gamma, beta, stats);

    const int n4 = (BB * OUTC * OUTL * TT) / 4;         // 2,097,152
    k_bn_prelu<<<n4 / 256, 256, 0, stream>>>(conv, stats, alpha, out);
}
